// MultiheadSelfAttention_57312043598369
// MI455X (gfx1250) — matrix operand order repaired and checked by reading, not compiled, not run on hardware
//
#include <hip/hip_runtime.h>
#include <hip/hip_bf16.h>

// ---------------- constants ----------------
#define BATCH 4
#define SEQ   2048
#define DMODEL 1024
#define NHEAD 16
#define DK    64
#define BLH   (BATCH * SEQ)            // 8192 rows
#define SCALE 0.125f                   // 1/sqrt(64)

typedef __attribute__((ext_vector_type(16))) __bf16 v16bf;
typedef __attribute__((ext_vector_type(8)))  float  v8f;

union Frag {
    v16bf v;
    uint4 q[2];
    unsigned int u[8];
};

__device__ __forceinline__ unsigned short bf16_rn(float f) {
    unsigned int u = __float_as_uint(f);
    unsigned int r = 0x7FFFu + ((u >> 16) & 1u);
    return (unsigned short)((u + r) >> 16);
}

__device__ __forceinline__ v8f wmma_bf16(v16bf a, v16bf b, v8f c) {
    // (neg_a, A, neg_b, B, c_mod, C, reuse_a, reuse_b)
    return __builtin_amdgcn_wmma_f32_16x16x32_bf16(false, a, false, b, (short)0, c, false, false);
}

// one per-lane b128 async copy global -> LDS (ASYNCcnt tracked)
__device__ __forceinline__ void async_b128_to_lds(unsigned ldsByteAddr, const void* gaddr) {
    asm volatile("global_load_async_to_lds_b128 %0, %1, off"
                 :: "v"(ldsByteAddr), "v"(gaddr) : "memory");
}

__device__ __forceinline__ unsigned lds_addr_of(const void* p) {
    return (unsigned)(unsigned long long)(__attribute__((address_space(3))) const char*)p;
}

// ---------------- f32 -> bf16 convert ----------------
__global__ void k_f32_to_bf16(const float* __restrict__ in, unsigned short* __restrict__ out, int n) {
    int i = blockIdx.x * blockDim.x + threadIdx.x;
    if (i < n) out[i] = bf16_rn(in[i]);
}

// ---------------- bf16 GEMM: C[M,N] = A[M,K] @ B[N,K]^T ----------------
// wave tile: 32 rows x 64 cols (2 A-frags share 4 B-frags -> 8 WMMA/K-step),
// 8 waves/block -> 256x64 block tile. B tile (64x32 bf16 = 4KB) async-copied
// to LDS and double-buffered; A frags register-double-buffered.
__global__ __launch_bounds__(256) void k_gemm_bf16(
    const unsigned short* __restrict__ A, const unsigned short* __restrict__ Bw,
    float* __restrict__ C, int M, int N, int K) {
    __shared__ __align__(16) unsigned short Btile[2][2048];   // 2 x (64 rows x 32 k)

    const int tid  = threadIdx.x;
    const int lane = tid & 31;
    const int wave = tid >> 5;
    const int m    = lane & 15;
    const bool hi  = lane >= 16;
    const int row0 = blockIdx.y * 256 + wave * 32;
    const int col0 = blockIdx.x * 64;

    const size_t arow0 = (size_t)(row0 + m) * K;
    const size_t arow1 = (size_t)(row0 + 16 + m) * K;
    // cooperative B-tile source: thread t loads row (t>>2), 8-element chunk (t&3)
    const unsigned short* bsrc = Bw + (size_t)(col0 + (tid >> 2)) * K + (size_t)((tid & 3) * 8);
    const unsigned ldsBase = lds_addr_of(&Btile[0][0]);
    const unsigned ldsMine = ldsBase + (unsigned)tid * 16u;

    v8f acc0[4] = {}, acc1[4] = {};
    Frag a_cur[2], a_nxt[2];

    // prologue: tile kb=0 into buf0, A frags kb=0 into regs
    async_b128_to_lds(ldsMine, bsrc);
    {
        const int klo = hi ? 8 : 0;       // A-frag: lane<16 K 0..7 & 16..23 ; lane>=16 8..15 & 24..31
        a_cur[0].q[0] = *(const uint4*)(A + arow0 + klo);
        a_cur[0].q[1] = *(const uint4*)(A + arow0 + klo + 16);
        a_cur[1].q[0] = *(const uint4*)(A + arow1 + klo);
        a_cur[1].q[1] = *(const uint4*)(A + arow1 + klo + 16);
    }

    int buf = 0;
    for (int kb = 0; kb < K; kb += 32) {
        const bool more = (kb + 32) < K;
        __syncthreads();                                  // prior reads of buf^1 done (WAR)
        if (more) {
            async_b128_to_lds(ldsMine + (unsigned)((buf ^ 1) * 4096), bsrc + (kb + 32));
            const int klo = kb + 32 + (hi ? 8 : 0);
            a_nxt[0].q[0] = *(const uint4*)(A + arow0 + klo);
            a_nxt[0].q[1] = *(const uint4*)(A + arow0 + klo + 16);
            a_nxt[1].q[0] = *(const uint4*)(A + arow1 + klo);
            a_nxt[1].q[1] = *(const uint4*)(A + arow1 + klo + 16);
            asm volatile("s_wait_asynccnt 0x1" ::: "memory");   // current buf landed, next in flight
        } else {
            asm volatile("s_wait_asynccnt 0x0" ::: "memory");
        }
        __syncthreads();                                  // tile visible to all waves
        const unsigned short* bt = &Btile[buf][0];
        const int kloc = hi ? 8 : 0;                      // B-frag: lane<16 K 0..7 & 16..23 ; lane>=16 8..15 & 24..31
        Frag bf[4];
#pragma unroll
        for (int c = 0; c < 4; ++c) {
            const int n = c * 16 + m;
            bf[c].q[0] = *(const uint4*)(bt + n * 32 + kloc);
            bf[c].q[1] = *(const uint4*)(bt + n * 32 + kloc + 16);
        }
#pragma unroll
        for (int c = 0; c < 4; ++c) {
            acc0[c] = wmma_bf16(a_cur[0].v, bf[c].v, acc0[c]);
            acc1[c] = wmma_bf16(a_cur[1].v, bf[c].v, acc1[c]);
        }
        a_cur[0] = a_nxt[0];
        a_cur[1] = a_nxt[1];
        buf ^= 1;
    }

    const int hi8 = hi ? 8 : 0;
#pragma unroll
    for (int c = 0; c < 4; ++c)
#pragma unroll
        for (int r = 0; r < 8; ++r) {
            C[(size_t)(row0 + r + hi8) * N + col0 + c * 16 + m]      = acc0[c][r];
            C[(size_t)(row0 + 16 + r + hi8) * N + col0 + c * 16 + m] = acc1[c][r];
        }
}

// ---------------- RoPE + split into heads: [B*L,D] f32 -> [B,H,L,dk] bf16 ----------------
__global__ void k_rope_heads(const float* __restrict__ P, const float* __restrict__ cosT,
                             const float* __restrict__ sinT, const int* __restrict__ pos,
                             unsigned short* __restrict__ out) {
    int i = blockIdx.x * blockDim.x + threadIdx.x;          // B*L*H*32 threads
    int j = i & 31;                                         // pair index (dk/2 = 32)
    int h = (i >> 5) & 15;
    int l = (i >> 9) & (SEQ - 1);
    int b = i >> 20;
    int p = pos[l];
    float c = cosT[p * 32 + j];
    float s = sinT[p * 32 + j];
    size_t src = ((size_t)(b * SEQ + l)) * DMODEL + h * DK + 2 * j;
    float t1 = P[src], t2 = P[src + 1];
    size_t dst = (((size_t)(b * NHEAD + h)) * SEQ + l) * DK + 2 * j;
    out[dst]     = bf16_rn(t1 * c - t2 * s);
    out[dst + 1] = bf16_rn(t1 * s + t2 * c);
}

// ---------------- V transpose: [B*L,D] f32 -> [B,H,dk,L] bf16 ----------------
__global__ void k_v_transpose(const float* __restrict__ P, unsigned short* __restrict__ Vt) {
    int i = blockIdx.x * blockDim.x + threadIdx.x;          // B*H*DK*L threads
    int l = i & (SEQ - 1);
    int d = (i >> 11) & 63;
    int h = (i >> 17) & 15;
    int b = i >> 21;
    float v = P[((size_t)(b * SEQ + l)) * DMODEL + h * DK + d];
    Vt[(((size_t)(b * NHEAD + h)) * DK + d) * SEQ + l] = bf16_rn(v);
}

// ---------------- causal flash attention ----------------
// Block: 4 waves x 16 q-rows = 64 q rows. K tile (32x64) and V tile (64x32)
// are async-double-buffered in LDS and shared by all waves. S^T = K @ Q^T is
// computed so the S^T C-layout maps 1:1 onto the A-operand layout of P.
__global__ __launch_bounds__(128) void k_attention(
    const unsigned short* __restrict__ Qh, const unsigned short* __restrict__ Kh,
    const unsigned short* __restrict__ Vt, unsigned short* __restrict__ Ao) {
    __shared__ __align__(16) unsigned short Kt[2][2048];    // 32 k-rows x 64 d
    __shared__ __align__(16) unsigned short Vl[2][2048];    // 64 d-rows x 32 k

    const int tid  = threadIdx.x;
    const int lane = tid & 31;
    const int wave = tid >> 5;
    const int m    = lane & 15;
    const bool hi  = lane >= 16;
    const int hi8  = hi ? 8 : 0;
    const int bh   = blockIdx.x;
    const int qtB  = blockIdx.y * 64;
    const int qt   = qtB + wave * 16;
    const int q    = qt + m;

    const unsigned short* Qb = Qh + (size_t)bh * SEQ * DK;
    const unsigned short* Kb = Kh + (size_t)bh * SEQ * DK;
    const unsigned short* Vb = Vt + (size_t)bh * DK * SEQ;

    const unsigned ktBase = lds_addr_of(&Kt[0][0]);
    const unsigned vtBase = lds_addr_of(&Vl[0][0]);
    const int c0 = tid * 2, c1 = tid * 2 + 1;               // 2 b128 chunks each for K and V

    // issue one 32-k tile (K rows kb..kb+31, V cols kb..kb+31) into buffer `bf2`
    auto issue_tile = [&](int bf2, int kb) {
        const unsigned kB = ktBase + (unsigned)(bf2 * 4096);
        const unsigned vB = vtBase + (unsigned)(bf2 * 4096);
        async_b128_to_lds(kB + (unsigned)c0 * 16u, Kb + (size_t)(kb + (c0 >> 3)) * DK + (c0 & 7) * 8);
        async_b128_to_lds(kB + (unsigned)c1 * 16u, Kb + (size_t)(kb + (c1 >> 3)) * DK + (c1 & 7) * 8);
        async_b128_to_lds(vB + (unsigned)c0 * 16u, Vb + (size_t)(c0 >> 2) * SEQ + kb + (c0 & 3) * 8);
        async_b128_to_lds(vB + (unsigned)c1 * 16u, Vb + (size_t)(c1 >> 2) * SEQ + kb + (c1 & 3) * 8);
    };

    // Q as B-operand: N = q row (lane&15), dk 8h..8h+7 & 16+8h..16+8h+7 per lane half
    Frag qf[2];
    {
        const size_t qrow = (size_t)(qt + m) * DK;
        const int dbase = hi ? 8 : 0;
        qf[0].q[0] = *(const uint4*)(Qb + qrow + dbase);
        qf[0].q[1] = *(const uint4*)(Qb + qrow + dbase + 16);
        qf[1].q[0] = *(const uint4*)(Qb + qrow + 32 + dbase);
        qf[1].q[1] = *(const uint4*)(Qb + qrow + 32 + dbase + 16);
    }

    v8f o[4] = {};
    float mrow = -1e30f, lrow = 0.0f;
    const int kend = qtB + 64;                  // block-uniform; causality via mask only

    issue_tile(0, 0);
    int buf = 0;
    for (int kb = 0; kb < kend; kb += 32) {
        const bool more = (kb + 32) < kend;
        __syncthreads();                        // prior reads of buf^1 done (WAR)
        if (more) {
            issue_tile(buf ^ 1, kb + 32);
            asm volatile("s_wait_asynccnt 0x4" ::: "memory");  // current 4 landed, next 4 in flight
        } else {
            asm volatile("s_wait_asynccnt 0x0" ::: "memory");
        }
        __syncthreads();                        // tile visible to all waves
        const unsigned short* ktp = &Kt[buf][0];
        const unsigned short* vtp = &Vl[buf][0];
        const int d0 = hi ? 8 : 0;

        // ---- S^T tiles: st0 = K[kb..kb+15] @ Q^T, st1 = K[kb+16..kb+31] @ Q^T ----
        Frag ka0, ka1;
        ka0.q[0] = *(const uint4*)(ktp + m * 64 + d0);
        ka0.q[1] = *(const uint4*)(ktp + m * 64 + d0 + 16);
        ka1.q[0] = *(const uint4*)(ktp + m * 64 + 32 + d0);
        ka1.q[1] = *(const uint4*)(ktp + m * 64 + 32 + d0 + 16);
        v8f st0 = {};
        st0 = wmma_bf16(ka0.v, qf[0].v, st0);
        st0 = wmma_bf16(ka1.v, qf[1].v, st0);
        ka0.q[0] = *(const uint4*)(ktp + (16 + m) * 64 + d0);
        ka0.q[1] = *(const uint4*)(ktp + (16 + m) * 64 + d0 + 16);
        ka1.q[0] = *(const uint4*)(ktp + (16 + m) * 64 + 32 + d0);
        ka1.q[1] = *(const uint4*)(ktp + (16 + m) * 64 + 32 + d0 + 16);
        v8f st1 = {};
        st1 = wmma_bf16(ka0.v, qf[0].v, st1);
        st1 = wmma_bf16(ka1.v, qf[1].v, st1);

        // ---- causal mask + scale ----
#pragma unroll
        for (int r = 0; r < 8; ++r) {
            const int k0 = kb + r + hi8;
            const int k1 = kb + 16 + r + hi8;
            st0[r] = (k0 <= q) ? st0[r] * SCALE : -1e30f;
            st1[r] = (k1 <= q) ? st1[r] * SCALE : -1e30f;
        }
        // ---- online softmax (per-lane row stats; merge lane halves) ----
        float rmax = -1e30f;
#pragma unroll
        for (int r = 0; r < 8; ++r) rmax = fmaxf(rmax, fmaxf(st0[r], st1[r]));
        rmax = fmaxf(rmax, __shfl_xor(rmax, 16, 32));
        const float mnew = fmaxf(mrow, rmax);
        const float corr = __expf(mrow - mnew);
        float rsum = 0.0f;
#pragma unroll
        for (int r = 0; r < 8; ++r) {
            st0[r] = __expf(st0[r] - mnew);
            st1[r] = __expf(st1[r] - mnew);
            rsum += st0[r] + st1[r];
        }
        rsum += __shfl_xor(rsum, 16, 32);
        lrow = lrow * corr + rsum;
        mrow = mnew;
        // ---- pack P into A-operand layout (S^T C-layout maps 1:1) ----
        Frag pa;
#pragma unroll
        for (int i2 = 0; i2 < 4; ++i2) {
            pa.u[i2]     = (unsigned int)bf16_rn(st0[2 * i2]) | ((unsigned int)bf16_rn(st0[2 * i2 + 1]) << 16);
            pa.u[4 + i2] = (unsigned int)bf16_rn(st1[2 * i2]) | ((unsigned int)bf16_rn(st1[2 * i2 + 1]) << 16);
        }
        // ---- O = O*corr + P @ V ----
        const int kloc = hi ? 8 : 0;
        Frag vf[4];
#pragma unroll
        for (int nb = 0; nb < 4; ++nb) {
            vf[nb].q[0] = *(const uint4*)(vtp + (nb * 16 + m) * 32 + kloc);
            vf[nb].q[1] = *(const uint4*)(vtp + (nb * 16 + m) * 32 + kloc + 16);
        }
#pragma unroll
        for (int nb = 0; nb < 4; ++nb)
#pragma unroll
            for (int r = 0; r < 8; ++r) o[nb][r] *= __shfl(corr, r + hi8, 32);
#pragma unroll
        for (int nb = 0; nb < 4; ++nb)
            o[nb] = wmma_bf16(pa.v, vf[nb].v, o[nb]);
        buf ^= 1;
    }
    // ---- finalize: divide by l, write bf16 [B*L, D] ----
    const float rl = 1.0f / lrow;
    const int b = bh / NHEAD;
    const int h = bh % NHEAD;
#pragma unroll
    for (int nb = 0; nb < 4; ++nb)
#pragma unroll
        for (int r = 0; r < 8; ++r) {
            const float val = o[nb][r] * __shfl(rl, r + hi8, 32);
            const int qrow = qt + r + hi8;
            Ao[((size_t)(b * SEQ + qrow)) * DMODEL + h * DK + nb * 16 + m] = bf16_rn(val);
        }
}

// ---------------- host launcher ----------------
extern "C" void kernel_launch(void* const* d_in, const int* in_sizes, int n_in,
                              void* d_out, int out_size, void* d_ws, size_t ws_size,
                              hipStream_t stream) {
    const float* x   = (const float*)d_in[0];
    const float* Wq  = (const float*)d_in[1];
    const float* Wk  = (const float*)d_in[2];
    const float* Wv  = (const float*)d_in[3];
    const float* Wo  = (const float*)d_in[4];
    const float* cosT = (const float*)d_in[5];
    const float* sinT = (const float*)d_in[6];
    const int*   pos  = (const int*)d_in[7];

    const size_t MB = 1024 * 1024;
    char* ws = (char*)d_ws;
    unsigned short* xb   = (unsigned short*)(ws + 0);        // 16 MB  (8M bf16)
    unsigned short* wqb  = (unsigned short*)(ws + 16 * MB);  // 2 MB
    unsigned short* wkb  = (unsigned short*)(ws + 18 * MB);
    unsigned short* wvb  = (unsigned short*)(ws + 20 * MB);
    unsigned short* wob  = (unsigned short*)(ws + 22 * MB);
    float*          Pf   = (float*)        (ws + 24 * MB);   // 32 MB scratch (reused q/k/v)
    unsigned short* Qh   = (unsigned short*)(ws + 56 * MB);  // 16 MB [B,H,L,dk]
    unsigned short* Kh   = (unsigned short*)(ws + 72 * MB);  // 16 MB
    unsigned short* Vt   = (unsigned short*)(ws + 88 * MB);  // 16 MB [B,H,dk,L]
    unsigned short* Ao   = (unsigned short*)(ws + 104 * MB); // 16 MB [B*L,D]

    const int nX = BLH * DMODEL;        // 8,388,608
    const int nW = DMODEL * DMODEL;     // 1,048,576

    k_f32_to_bf16<<<nX / 256, 256, 0, stream>>>(x,  xb,  nX);
    k_f32_to_bf16<<<nW / 256, 256, 0, stream>>>(Wq, wqb, nW);
    k_f32_to_bf16<<<nW / 256, 256, 0, stream>>>(Wk, wkb, nW);
    k_f32_to_bf16<<<nW / 256, 256, 0, stream>>>(Wv, wvb, nW);
    k_f32_to_bf16<<<nW / 256, 256, 0, stream>>>(Wo, wob, nW);

    const dim3 gGemm(DMODEL / 64, BLH / 256);   // (16, 32)
    const int nRope = BATCH * SEQ * NHEAD * (DK / 2);  // 4,194,304

    // Q = x @ Wq^T -> RoPE -> heads
    k_gemm_bf16<<<gGemm, 256, 0, stream>>>(xb, wqb, Pf, BLH, DMODEL, DMODEL);
    k_rope_heads<<<nRope / 256, 256, 0, stream>>>(Pf, cosT, sinT, pos, Qh);
    // K = x @ Wk^T -> RoPE -> heads
    k_gemm_bf16<<<gGemm, 256, 0, stream>>>(xb, wkb, Pf, BLH, DMODEL, DMODEL);
    k_rope_heads<<<nRope / 256, 256, 0, stream>>>(Pf, cosT, sinT, pos, Kh);
    // V = x @ Wv^T -> transposed heads
    k_gemm_bf16<<<gGemm, 256, 0, stream>>>(xb, wvb, Pf, BLH, DMODEL, DMODEL);
    k_v_transpose<<<nX / 256, 256, 0, stream>>>(Pf, Vt);

    // attention: 64 (b,h) x 32 q-blocks of 64 rows, 4 waves/block
    k_attention<<<dim3(BATCH * NHEAD, SEQ / 64), 128, 0, stream>>>(Qh, Kh, Vt, Ao);

    // out = attn @ Wo^T (f32 output)
    k_gemm_bf16<<<gGemm, 256, 0, stream>>>(Ao, wob, (float*)d_out, BLH, DMODEL, DMODEL);
}
